// STGCN_LSTM_Node_45174466019853
// MI455X (gfx1250) — compile-verified, matrix operand order checked by reading, not run on hardware
//
#include <hip/hip_runtime.h>
#include <hip/hip_bf16.h>
#include <cstdint>
#include <cstddef>

#define T_STEPS 128
#define NN      32
#define NE      512
#define INC     64
#define HID     128
#define G4      512      // 4*HID
#define SEQL    16
#define BT      16       // batch-tile (windows per workgroup)
#define HSP     136      // padded LDS row pitch in halves (272B: 68 dwords -> 4-bank stagger)

typedef _Float16 v16h __attribute__((ext_vector_type(16)));
typedef float    v8f  __attribute__((ext_vector_type(8)));
typedef unsigned int u32x4 __attribute__((ext_vector_type(4)));
typedef int          i32x4 __attribute__((ext_vector_type(4)));
typedef int          i32x8 __attribute__((ext_vector_type(8)));

#if defined(__HIP_DEVICE_COMPILE__) && __has_builtin(__builtin_amdgcn_tensor_load_to_lds)
#define HAVE_TDM 1
#else
#define HAVE_TDM 0
#endif

__device__ __forceinline__ float fsig(float x)  { return 1.f / (1.f + __expf(-x)); }
__device__ __forceinline__ float ftanh(float x) { float e = __expf(2.f * x); return (e - 1.f) / (e + 1.f); }

// ---------------- prep: small weight transposes + bias sums ----------------
__global__ void prep_small(const float* __restrict__ W1, const float* __restrict__ W2,
                           const float* __restrict__ bih1, const float* __restrict__ bhh1,
                           const float* __restrict__ bih2, const float* __restrict__ bhh2,
                           _Float16* __restrict__ w1t, _Float16* __restrict__ w2t,
                           float* __restrict__ bsum1, float* __restrict__ bsum2)
{
  int gid = blockIdx.x * blockDim.x + threadIdx.x;
  int stride = gridDim.x * blockDim.x;
  for (int i = gid; i < HID * INC; i += stride) {        // w1t[nn][k] = W1[k][nn]
    int nn = i / INC, k = i % INC;
    w1t[i] = (_Float16)W1[k * HID + nn];
  }
  for (int i = gid; i < HID * HID; i += stride) {        // w2t[nn][k] = W2[k][nn]
    int nn = i >> 7, k = i & 127;
    w2t[i] = (_Float16)W2[k * HID + nn];
  }
  for (int i = gid; i < NN * G4; i += stride) {
    bsum1[i] = bih1[i] + bhh1[i];
    bsum2[i] = bih2[i] + bhh2[i];
  }
}

// ---------------- prep: stacked [Wih;Whh] -> f16, per node/layer ----------------
__global__ void prep_wc(const float* __restrict__ Wih1, const float* __restrict__ Whh1,
                        const float* __restrict__ Wih2, const float* __restrict__ Whh2,
                        _Float16* __restrict__ wct)
{
  size_t id = (size_t)blockIdx.x * blockDim.x + threadIdx.x;
  const size_t perLayer = (size_t)NN * G4 * 256;
  if (id >= 2 * perLayer) return;
  int layer = id >= perLayer;
  size_t r = id - (size_t)layer * perLayer;
  int n = (int)(r >> 17);
  int g = (int)((r >> 8) & (G4 - 1));
  int k = (int)(r & 255);
  const float* wih = layer ? Wih2 : Wih1;
  const float* whh = layer ? Whh2 : Whh1;
  float v = (k < HID) ? wih[((size_t)n * G4 + g) * HID + k]
                      : whh[((size_t)n * G4 + g) * HID + (k - HID)];
  wct[id] = (_Float16)v;
}

// ---------------- GCN: one block per timestep ----------------
__global__ __launch_bounds__(128) void gcn_kernel(
    const float* __restrict__ x, const int* __restrict__ ei,
    const float* __restrict__ b1, const float* __restrict__ b2,
    const _Float16* __restrict__ w1t, const _Float16* __restrict__ w2t,
    _Float16* __restrict__ hgcn)
{
  __shared__ _Float16 xs[NN][72];
  __shared__ float    xw[NN][HID];
  __shared__ float    anorm[NN][NN + 1];
  __shared__ float    deg[NN];
  __shared__ _Float16 hs[NN][HSP];

  const int t = blockIdx.x;
  const int tid = threadIdx.x;
  const int wave = tid >> 5, lane = tid & 31;
  const int m = lane & 15, khalf = lane >> 4;

  const float* xt = x + (size_t)t * NN * INC;
  for (int i = tid; i < NN * INC; i += 128) xs[i / INC][i % INC] = (_Float16)xt[i];
  if (tid < NN) deg[tid] = 1.f;
  for (int i = tid; i < NN * (NN + 1); i += 128) (&anorm[0][0])[i] = 0.f;
  __syncthreads();

  const int* rowp = ei + (size_t)t * 2 * NE;
  const int* colp = rowp + NE;
  for (int e = tid; e < NE; e += 128) atomicAdd(&deg[colp[e]], 1.f);
  __syncthreads();
  if (tid < NN) deg[tid] = rsqrtf(deg[tid]);
  __syncthreads();
  if (tid < NN) atomicAdd(&anorm[tid][tid], deg[tid] * deg[tid]);
  for (int e = tid; e < NE; e += 128) {
    int rr = rowp[e], cc = colp[e];
    atomicAdd(&anorm[cc][rr], deg[rr] * deg[cc]);
  }
  __syncthreads();

  // layer 1: xw = xs[32,64] @ W1
  {
    const int mtile = wave & 1;
    const int nbase = (wave >> 1) * 4;
    v8f acc[4] = {};
    for (int kc = 0; kc < 2; ++kc) {
      v16h a;
      const _Float16* ap = &xs[mtile * 16 + m][kc * 32 + khalf * 8];
      ((float4*)&a)[0] = *(const float4*)ap;
      ((float4*)&a)[1] = *(const float4*)(ap + 16);
      #pragma unroll
      for (int i = 0; i < 4; ++i) {
        int g = (nbase + i) * 16 + m;
        const _Float16* bp = w1t + (size_t)g * INC + kc * 32 + khalf * 8;
        v16h bf;
        ((float4*)&bf)[0] = *(const float4*)bp;
        ((float4*)&bf)[1] = *(const float4*)(bp + 16);
        acc[i] = __builtin_amdgcn_wmma_f32_16x16x32_f16(false, a, false, bf,
                                                        (short)0, acc[i], false, false);
      }
    }
    #pragma unroll
    for (int i = 0; i < 4; ++i)
      #pragma unroll
      for (int r = 0; r < 8; ++r)
        xw[mtile * 16 + r + 8 * khalf][(nbase + i) * 16 + m] = acc[i][r];
  }
  __syncthreads();

  for (int c = tid; c < NN * HID; c += 128) {
    int nrow = c >> 7, h = c & 127;
    float s = b1[h];
    #pragma unroll 8
    for (int r = 0; r < NN; ++r) s += anorm[nrow][r] * xw[r][h];
    s = s > 0.f ? s : 0.f;
    hs[nrow][h] = (_Float16)s;
  }
  __syncthreads();

  // layer 2: xw = hs[32,128] @ W2
  {
    const int mtile = wave & 1;
    const int nbase = (wave >> 1) * 4;
    v8f acc[4] = {};
    for (int kc = 0; kc < 4; ++kc) {
      v16h a;
      const _Float16* ap = &hs[mtile * 16 + m][kc * 32 + khalf * 8];
      ((float4*)&a)[0] = *(const float4*)ap;
      ((float4*)&a)[1] = *(const float4*)(ap + 16);
      #pragma unroll
      for (int i = 0; i < 4; ++i) {
        int g = (nbase + i) * 16 + m;
        const _Float16* bp = w2t + (size_t)g * HID + kc * 32 + khalf * 8;
        v16h bf;
        ((float4*)&bf)[0] = *(const float4*)bp;
        ((float4*)&bf)[1] = *(const float4*)(bp + 16);
        acc[i] = __builtin_amdgcn_wmma_f32_16x16x32_f16(false, a, false, bf,
                                                        (short)0, acc[i], false, false);
      }
    }
    __syncthreads();
    #pragma unroll
    for (int i = 0; i < 4; ++i)
      #pragma unroll
      for (int r = 0; r < 8; ++r)
        xw[mtile * 16 + r + 8 * khalf][(nbase + i) * 16 + m] = acc[i][r];
  }
  __syncthreads();

  _Float16* hout = hgcn + (size_t)t * NN * HID;
  for (int c = tid; c < NN * HID; c += 128) {
    int nrow = c >> 7, h = c & 127;
    float s = b2[h];
    #pragma unroll 8
    for (int r = 0; r < NN; ++r) s += anorm[nrow][r] * xw[r][h];
    s = s > 0.f ? s : 0.f;
    hout[nrow * HID + h] = (_Float16)s;
  }
}

// ---------------- fused 2-layer LSTM + fc + heads ----------------
// grid: (node n = 0..31, batch-tile bt = 0..7), 256 threads = 8 waves
__global__ __launch_bounds__(256) void lstm_fused(
    const _Float16* __restrict__ hgcn,
    const _Float16* __restrict__ wc1t, const _Float16* __restrict__ wc2t,
    const float* __restrict__ bsum1, const float* __restrict__ bsum2,
    const float* __restrict__ Wfc,  const float* __restrict__ bfc,
    const float* __restrict__ Wout0, const float* __restrict__ bout0,
    const float* __restrict__ Wout1, const float* __restrict__ bout1,
    float* __restrict__ out)
{
  extern __shared__ char smem[];
  _Float16* h1s    = (_Float16*)smem;              // [SEQL*BT][HSP] layer-1 h outputs
  _Float16* hstage = h1s + SEQL * BT * HSP;        // [32][HSP] staged window inputs
  _Float16* xh     = hstage + 32 * HSP;            // [BT][HSP]  h-state
  float*    gbuf   = (float*)(xh + BT * HSP);      // [BT][G4]   gate accum / fc out
  float*    cst    = gbuf + BT * G4;               // [BT][HID]  c-state / relu buf

  const int n  = blockIdx.x;
  const int bt = blockIdx.y;
  const int tid = threadIdx.x;
  const int wave = tid >> 5, lane = tid & 31;
  const int m = lane & 15, khalf = lane >> 4;

  const int first_t = bt * BT - (SEQL - 1);                 // window union: 31 rows
  const int skip    = (first_t < 0) ? -first_t : 0;
  const int nrows   = 31 - skip;

  for (int i = tid; i < BT * HID; i += 256) {
    cst[i] = 0.f;
    xh[(i >> 7) * HSP + (i & 127)] = (_Float16)0.f;
  }

#if HAVE_TDM
  // zero the front-padded (t<0) rows; TDM fills the valid rows asynchronously
  for (int i = tid; i < skip * (HID / 8); i += 256) {
    int j = i >> 4, h8 = (i & 15) * 8;
    *(float4*)(hstage + j * HSP + h8) = make_float4(0.f, 0.f, 0.f, 0.f);
  }
  if (wave == 0) {
    unsigned long long gaddr =
        (unsigned long long)(uintptr_t)(hgcn + ((size_t)(first_t + skip) * NN + n) * HID);
    unsigned int laddr = (unsigned int)(uintptr_t)(hstage + skip * HSP);
    // D# group 0: count=1, lds_addr, global_addr[56:0], type=2
    u32x4 g0 = { 1u,
                 laddr,
                 (unsigned int)(gaddr & 0xFFFFFFFFu),
                 (unsigned int)((gaddr >> 32) & 0x01FFFFFFu) | (2u << 30) };
    // D# group 1: data_size=2B, pad_enable, pad_interval=64dw (code 5), pad_amount=4dw (code 3)
    // tensor_dim0=128, tensor_dim1=nrows, tile_dim0=128, tile_dim1=nrows, dim0_stride=NN*HID
    i32x8 g1 = { (int)((1u << 16) | (1u << 20) | (5u << 22) | (3u << 25)),
                 (int)((HID & 0xFFFF) << 16),
                 (int)(((unsigned)nrows & 0xFFFFu) << 16),
                 (int)(HID << 16),
                 (int)((unsigned)nrows & 0xFFFFu),
                 (int)(NN * HID),
                 0, 0 };
    i32x4 z4 = { 0, 0, 0, 0 };
#if __has_include(<hip/amd_detail/amd_gfx1250_TDM.h>)
    i32x8 z8 = { 0, 0, 0, 0, 0, 0, 0, 0 };
    __builtin_amdgcn_tensor_load_to_lds(g0, g1, z4, z4, z8, 0);   // 6-arg (clang-23/therock)
#else
    __builtin_amdgcn_tensor_load_to_lds(g0, g1, z4, z4, 0);       // 5-arg (ROCm 7.2)
#endif
    __builtin_amdgcn_s_wait_tensorcnt(0);
  }
#else
  // fallback: cooperative b128 staging
  for (int i = tid; i < 31 * (HID / 8); i += 256) {
    int j = i >> 4, h8 = (i & 15) * 8;
    int tt = first_t + j;
    float4 v = make_float4(0.f, 0.f, 0.f, 0.f);
    if (tt >= 0) v = *(const float4*)(hgcn + ((size_t)tt * NN + n) * HID + h8);
    *(float4*)(hstage + j * HSP + h8) = v;
  }
#endif
  __syncthreads();

  for (int layer = 0; layer < 2; ++layer) {
    const _Float16* wcn = (layer ? wc2t : wc1t) + (size_t)n * G4 * 256;
    const float* bs = (layer ? bsum2 : bsum1) + n * G4;

    for (int l = 0; l < SEQL; ++l) {
      // ---- gates[16,512] = [x_l ; h] (16x256) @ WcT^T via WMMA ----
      // A-fragments: kc<4 read the step input in place (hstage / h1s), kc>=4 read h-state.
      {
        const int nbase = wave * 4;
        const _Float16* inrow = (layer == 0) ? (hstage + (m + l) * HSP)
                                             : (h1s + (l * BT + m) * HSP);
        const _Float16* strow = xh + m * HSP;
        v8f acc[4] = {};
        #pragma unroll
        for (int kc = 0; kc < 8; ++kc) {
          const _Float16* ap = (kc < 4) ? (inrow + kc * 32 + khalf * 8)
                                        : (strow + (kc - 4) * 32 + khalf * 8);
          v16h a;
          ((float4*)&a)[0] = *(const float4*)ap;
          ((float4*)&a)[1] = *(const float4*)(ap + 16);
          #pragma unroll
          for (int i = 0; i < 4; ++i) {
            int g = (nbase + i) * 16 + m;
            const _Float16* bp = wcn + (size_t)g * 256 + kc * 32 + khalf * 8;
            v16h bf;
            ((float4*)&bf)[0] = *(const float4*)bp;
            ((float4*)&bf)[1] = *(const float4*)(bp + 16);
            acc[i] = __builtin_amdgcn_wmma_f32_16x16x32_f16(false, a, false, bf,
                                                            (short)0, acc[i], false, false);
          }
        }
        #pragma unroll
        for (int i = 0; i < 4; ++i)
          #pragma unroll
          for (int r = 0; r < 8; ++r)
            gbuf[(r + 8 * khalf) * G4 + (nbase + i) * 16 + m] = acc[i][r];
      }
      __syncthreads();

      // ---- elementwise LSTM cell (gate order i,f,g,o) ----
      for (int c = tid; c < BT * HID; c += 256) {
        int b = c >> 7, h = c & 127;
        float gi = gbuf[b * G4 + h]           + bs[h];
        float gf = gbuf[b * G4 + HID + h]     + bs[HID + h];
        float gg = gbuf[b * G4 + 2 * HID + h] + bs[2 * HID + h];
        float go = gbuf[b * G4 + 3 * HID + h] + bs[3 * HID + h];
        float cc = fsig(gf) * cst[c] + fsig(gi) * ftanh(gg);
        cst[c] = cc;
        float hh = fsig(go) * ftanh(cc);
        xh[b * HSP + h] = (_Float16)hh;
        if (layer == 0) h1s[(l * BT + b) * HSP + h] = (_Float16)hh;
      }
      __syncthreads();
    }

    if (layer == 0) {   // reset state for layer 2
      for (int i = tid; i < BT * HID; i += 256) {
        cst[i] = 0.f;
        xh[(i >> 7) * HSP + (i & 127)] = (_Float16)0.f;
      }
      __syncthreads();
    }
  }

  // ---- last = relu(h2 at final step) -> cst (f32) ----
  for (int i = tid; i < BT * HID; i += 256) {
    float v = (float)xh[(i >> 7) * HSP + (i & 127)];
    cst[i] = v > 0.f ? v : 0.f;
  }
  __syncthreads();

  // ---- fc ----
  const float* wfcn = Wfc + (size_t)n * HID * HID;
  for (int c = tid; c < BT * HID; c += 256) {
    int b = c >> 7, k = c & 127;
    const float* w  = wfcn + k * HID;
    const float* xx = cst + b * HID;
    float s = 0.f;
    #pragma unroll 4
    for (int h = 0; h < HID; ++h) s += xx[h] * w[h];
    gbuf[b * HID + k] = s + bfc[n * HID + k];
  }
  __syncthreads();

  // ---- heads, transposed writes: out0 [T][N][4], out1 [T][N][2] ----
  if (tid < BT * 4) {
    int b = tid >> 2, k = tid & 3;
    const float* w  = Wout0 + ((size_t)n * 4 + k) * HID;
    const float* xx = gbuf + b * HID;
    float s = 0.f;
    for (int h = 0; h < HID; ++h) s += xx[h] * w[h];
    int tt = bt * BT + b;
    out[((size_t)tt * NN + n) * 4 + k] = s + bout0[n * 4 + k];
  } else if (tid >= 64 && tid < 64 + BT * 2) {
    int q = tid - 64;
    int b = q >> 1, k = q & 1;
    const float* w  = Wout1 + ((size_t)n * 2 + k) * HID;
    const float* xx = gbuf + b * HID;
    float s = 0.f;
    for (int h = 0; h < HID; ++h) s += xx[h] * w[h];
    int tt = bt * BT + b;
    out[(size_t)T_STEPS * NN * 4 + ((size_t)tt * NN + n) * 2 + k] = s + bout1[n * 2 + k];
  }
}

// ---------------- host ----------------
extern "C" void kernel_launch(void* const* d_in, const int* in_sizes, int n_in,
                              void* d_out, int out_size, void* d_ws, size_t ws_size,
                              hipStream_t stream)
{
  const float* x    = (const float*)d_in[0];
  const int*   ei   = (const int*)d_in[1];
  const float* y    = (const float*)d_in[3];
  const float* W1   = (const float*)d_in[4];
  const float* b1   = (const float*)d_in[5];
  const float* W2   = (const float*)d_in[6];
  const float* b2   = (const float*)d_in[7];
  const float* Wih1 = (const float*)d_in[8];
  const float* Whh1 = (const float*)d_in[9];
  const float* bih1 = (const float*)d_in[10];
  const float* bhh1 = (const float*)d_in[11];
  const float* Wih2 = (const float*)d_in[12];
  const float* Whh2 = (const float*)d_in[13];
  const float* bih2 = (const float*)d_in[14];
  const float* bhh2 = (const float*)d_in[15];
  const float* Wfc  = (const float*)d_in[16];
  const float* bfc  = (const float*)d_in[17];
  const float* Wout0 = (const float*)d_in[18];
  const float* bout0 = (const float*)d_in[19];
  const float* Wout1 = (const float*)d_in[20];
  const float* bout1 = (const float*)d_in[21];

  char* ws = (char*)d_ws;
  size_t off = 0;
  auto alloc = [&](size_t bytes) { size_t p = off; off = (off + bytes + 255) & ~(size_t)255; return p; };
  _Float16* w1t   = (_Float16*)(ws + alloc((size_t)HID * INC * 2));
  _Float16* w2t   = (_Float16*)(ws + alloc((size_t)HID * HID * 2));
  float*    bsum1 = (float*)(ws + alloc((size_t)NN * G4 * 4));
  float*    bsum2 = (float*)(ws + alloc((size_t)NN * G4 * 4));
  _Float16* hgcn  = (_Float16*)(ws + alloc((size_t)T_STEPS * NN * HID * 2));
  _Float16* wct   = (_Float16*)(ws + alloc((size_t)2 * NN * G4 * 256 * 2));

  prep_small<<<64, 256, 0, stream>>>(W1, W2, bih1, bhh1, bih2, bhh2, w1t, w2t, bsum1, bsum2);

  const size_t wcElems = (size_t)2 * NN * G4 * 256;
  prep_wc<<<(unsigned)((wcElems + 255) / 256), 256, 0, stream>>>(Wih1, Whh1, Wih2, Whh2, wct);

  gcn_kernel<<<T_STEPS, 128, 0, stream>>>(x, ei, b1, b2, w1t, w2t, hgcn);

  const int smemBytes = (SEQL * BT * HSP + 32 * HSP + BT * HSP) * 2   // h1s + hstage + xh
                      + BT * G4 * 4                                   // gbuf
                      + BT * HID * 4;                                 // cst (~121 KB)
  hipFuncSetAttribute(reinterpret_cast<const void*>(lstm_fused),
                      hipFuncAttributeMaxDynamicSharedMemorySize, smemBytes);

  lstm_fused<<<dim3(NN, T_STEPS / BT), 256, smemBytes, stream>>>(
      hgcn, wct, wct + (size_t)NN * G4 * 256, bsum1, bsum2,
      Wfc, bfc, Wout0, bout0, Wout1, bout1, (float*)d_out);

  hipMemcpyAsync((float*)d_out + (size_t)T_STEPS * NN * (4 + 2), y,
                 (size_t)T_STEPS * NN * 4 * sizeof(float),
                 hipMemcpyDeviceToDevice, stream);
}
